// Moe_43456479101230
// MI455X (gfx1250) — matrix operand order repaired and checked by reading, not compiled, not run on hardware
//
#include <hip/hip_runtime.h>
#include <hip/hip_bf16.h>
#include <math.h>

// ---------------- CDNA5 WMMA types ----------------
typedef __attribute__((ext_vector_type(16))) __bf16 v16bf;
typedef __attribute__((ext_vector_type(8)))  float  v8f;

union Frag {
  uint4 u4[2];
  v16bf v;
};

__device__ __forceinline__ unsigned short f2bf(float f) {
  union { float f; unsigned u; } x; x.f = f;
  unsigned u = x.u;
  return (unsigned short)((u + 0x7FFFu + ((u >> 16) & 1u)) >> 16);  // RNE
}

__device__ __forceinline__ uint4 pack8(const float f[8]) {
  uint4 pk;
  pk.x = (unsigned)f2bf(f[0]) | ((unsigned)f2bf(f[1]) << 16);
  pk.y = (unsigned)f2bf(f[2]) | ((unsigned)f2bf(f[3]) << 16);
  pk.z = (unsigned)f2bf(f[4]) | ((unsigned)f2bf(f[5]) << 16);
  pk.w = (unsigned)f2bf(f[6]) | ((unsigned)f2bf(f[7]) << 16);
  return pk;
}

__device__ __forceinline__ float gelu_erf(float x) {
  return 0.5f * x * (1.0f + erff(x * 0.70710678118654752f));
}

// ---------------- Generic WMMA GEMM ----------------
// D[M,N] = act(alpha*A@B + bias) * rowscale + addend.
// Block tile 64(M) x 128(N), 8 waves, 4 WMMA accumulators per wave.
struct GemmParams {
  const float* A; const float* B; float* C;
  const float* bias; const float* addend; const float* rowscale;
  const int* sel;                 // per-batch expert selector (nullable)
  long strideA, strideB, strideC, selStrideB;
  int lda, ldb, ldc;              // ldc: row stride (store_img=0) or col stride S (store_img=1)
  int M, N, K;
  float alpha;
  int amode;      // 0=row-major A, 1=3x3 SAME conv gather ([B,C,196]), 2=patch16 gather (x[B,3,224,224])
  int bmode;      // 0=W[N,K] row-major, 1=B[K,N] row-major
  int act;        // 0=none, 1=gelu(erf)
  int store_img;  // 0: C[row*ldc+col], 1: C[col*ldc+row]
  int biasSel;    // bias += sel[z]*N
  int splitk;     // 0: z = batch. >0: z = K-chunk of this many elements (partial store)
};

struct BRegs { uint4 p0, p1; unsigned short s[16]; };

template <int AMODE, int BMODE>
__launch_bounds__(256)
__global__ void gemm_k(GemmParams p) {
  __shared__ __align__(16) unsigned short As[64][40];   // [m][k] (+pad, 16B-aligned rows)
  __shared__ __align__(16) unsigned short Bs[128][40];  // [n][k]

  const int tid  = threadIdx.x;
  const int lane = tid & 31;
  const int wv   = tid >> 5;
  const int wrow = wv & 3;       // 4 row tiles of 16
  const int wcol = wv >> 2;      // 2 col halves of 64
  const int lr   = lane & 15;
  const int hf   = lane >> 4;
  const int z    = blockIdx.z;

  const int mBase = blockIdx.y * 64;
  const int nBase = blockIdx.x * 128;

  long aOff, bOff;
  int kBegin, kEnd;
  if (p.splitk > 0) {
    aOff = 0; bOff = 0;
    kBegin = z * p.splitk;
    kEnd = min(p.K, kBegin + p.splitk);
  } else {
    aOff = (long)z * p.strideA;
    bOff = p.sel ? (long)p.sel[z] * p.selStrideB : (long)z * p.strideB;
    kBegin = 0; kEnd = p.K;
  }
  const float* Bg = p.B + bOff;

  // ---- loop-invariant staging info ----
  const int am = tid >> 2;               // A row / B row0 within tile
  const int ak = (tid & 3) * 8;          // k offset of this thread's 8-run
  const int bk1 = tid >> 3;              // BMODE1: k within tile
  const int bn1 = (tid & 7) * 16;        // BMODE1: n-run of 16

  const int gmA = mBase + am;
  const bool rowOkA = gmA < p.M;
  const float* aRow;
  int py = 0, px = 0;
  if (AMODE == 0) aRow = p.A + aOff + (long)gmA * p.lda;
  else { py = gmA / 14; px = gmA - py * 14; aRow = p.A + aOff; }

  const int gnB0 = nBase + am, gnB1 = nBase + am + 64;
  const float* bRow0 = Bg + (long)gnB0 * p.ldb;
  const float* bRow1 = Bg + (long)gnB1 * p.ldb;
  const bool nOk0 = gnB0 < p.N, nOk1 = gnB1 < p.N;
  const bool nFast1 = (nBase + bn1 + 16) <= p.N;       // BMODE1 vector path

  auto loadA = [&](int k0) -> uint4 {
    float f[8];
    if (AMODE == 0) {
      if (rowOkA && (k0 + 8) <= kEnd) {
        const float4* s = (const float4*)(aRow + k0);
        float4 a0 = s[0], a1 = s[1];
        f[0]=a0.x; f[1]=a0.y; f[2]=a0.z; f[3]=a0.w;
        f[4]=a1.x; f[5]=a1.y; f[6]=a1.z; f[7]=a1.w;
      } else {
#pragma unroll
        for (int i = 0; i < 8; ++i) {
          const int gk = k0 + i;
          f[i] = (rowOkA && gk < kEnd) ? aRow[gk] : 0.f;
        }
      }
    } else if (AMODE == 1) {             // 3x3 SAME conv on 14x14, k=(cin,ky,kx)
      int cin = k0 / 9;
      int r = k0 - cin * 9;
      int ky = (r >= 6) ? 2 : (r >= 3 ? 1 : 0);
      int kx = r - ky * 3;
#pragma unroll
      for (int i = 0; i < 8; ++i) {
        const int iy = py + ky - 1, ix = px + kx - 1;
        const bool ok = rowOkA && (k0 + i) < kEnd &&
                        iy >= 0 && iy < 14 && ix >= 0 && ix < 14;
        f[i] = ok ? aRow[(long)cin * 196 + iy * 14 + ix] : 0.f;
        if (++kx == 3) { kx = 0; if (++ky == 3) { ky = 0; ++cin; } }
      }
    } else {                              // patch-embed gather, k=(c,i16,j16)
#pragma unroll
      for (int i = 0; i < 8; ++i) {
        const int gk = k0 + i;
        float v = 0.f;
        if (rowOkA && gk < kEnd) {
          const int c = gk >> 8, rem = gk & 255;
          const int r16 = rem >> 4, c16 = rem & 15;
          v = aRow[(long)(c * 224 + py * 16 + r16) * 224 + px * 16 + c16];
        }
        f[i] = v;
      }
    }
    return pack8(f);
  };

  auto loadB = [&](int k0) -> BRegs {
    BRegs r;
    if (BMODE == 0) {
      float f[8];
      if (nOk0 && (k0 + 8) <= kEnd) {
        const float4* s = (const float4*)(bRow0 + k0);
        float4 a0 = s[0], a1 = s[1];
        f[0]=a0.x; f[1]=a0.y; f[2]=a0.z; f[3]=a0.w;
        f[4]=a1.x; f[5]=a1.y; f[6]=a1.z; f[7]=a1.w;
      } else {
#pragma unroll
        for (int i = 0; i < 8; ++i) {
          const int gk = k0 + i;
          f[i] = (nOk0 && gk < kEnd) ? bRow0[gk] : 0.f;
        }
      }
      r.p0 = pack8(f);
      if (nOk1 && (k0 + 8) <= kEnd) {
        const float4* s = (const float4*)(bRow1 + k0);
        float4 a0 = s[0], a1 = s[1];
        f[0]=a0.x; f[1]=a0.y; f[2]=a0.z; f[3]=a0.w;
        f[4]=a1.x; f[5]=a1.y; f[6]=a1.z; f[7]=a1.w;
      } else {
#pragma unroll
        for (int i = 0; i < 8; ++i) {
          const int gk = k0 + i;
          f[i] = (nOk1 && gk < kEnd) ? bRow1[gk] : 0.f;
        }
      }
      r.p1 = pack8(f);
    } else {                              // B[K,N]: transpose-stage 16 n's at fixed k
      const int gk = k0 + bk1;
      const float* src = Bg + (long)gk * p.ldb + nBase + bn1;
      if (gk < kEnd && nFast1) {
        const float4* s4 = (const float4*)src;
#pragma unroll
        for (int q = 0; q < 4; ++q) {
          float4 v4 = s4[q];
          r.s[q * 4 + 0] = f2bf(v4.x); r.s[q * 4 + 1] = f2bf(v4.y);
          r.s[q * 4 + 2] = f2bf(v4.z); r.s[q * 4 + 3] = f2bf(v4.w);
        }
      } else {
#pragma unroll
        for (int j = 0; j < 16; ++j) {
          const int gn = nBase + bn1 + j;
          r.s[j] = (gk < kEnd && gn < p.N) ? f2bf(src[j]) : (unsigned short)0;
        }
      }
    }
    return r;
  };

  auto storeStage = [&](const uint4& ra, const BRegs& rb) {
    *(uint4*)&As[am][ak] = ra;
    if (BMODE == 0) {
      *(uint4*)&Bs[am][ak] = rb.p0;
      *(uint4*)&Bs[am + 64][ak] = rb.p1;
    } else {
#pragma unroll
      for (int j = 0; j < 16; ++j) Bs[bn1 + j][bk1] = rb.s[j];
    }
  };

  v8f acc[4] = {v8f{}, v8f{}, v8f{}, v8f{}};

  // ---- software-pipelined K loop ----
  {
    uint4 ra = loadA(kBegin);
    BRegs rb = loadB(kBegin);
    storeStage(ra, rb);
  }
  __syncthreads();

  for (int kk = kBegin; kk < kEnd; kk += 32) {
    const bool hasNext = (kk + 32) < kEnd;
    uint4 ra; BRegs rb;
    if (hasNext) { ra = loadA(kk + 32); rb = loadB(kk + 32); }

    Frag a;
    {
      const uint4* ap = (const uint4*)&As[16 * wrow + lr][hf * 8];
      a.u4[0] = ap[0]; a.u4[1] = ap[2];   // K {0..7,16..23} / {8..15,24..31}
    }
#pragma unroll
    for (int t = 0; t < 4; ++t) {
      Frag b;
      const uint4* bp = (const uint4*)&Bs[64 * wcol + 16 * t + lr][hf * 8];
      b.u4[0] = bp[0]; b.u4[1] = bp[2];   // K {0..7,16..23} / {8..15,24..31}
      acc[t] = __builtin_amdgcn_wmma_f32_16x16x32_bf16(false, a.v, false, b.v,
                                                       (short)0, acc[t], false, false);
    }
    __syncthreads();
    if (hasNext) storeStage(ra, rb);
    __syncthreads();
  }

  // ---- epilogue ----
  const float* biasp =
      p.bias ? (p.bias + ((p.biasSel && p.sel) ? (long)p.sel[z] * p.N : 0)) : nullptr;
  const float rs = p.rowscale ? p.rowscale[z] : 1.f;
  const long cOff = (long)z * p.strideC;
#pragma unroll
  for (int t = 0; t < 4; ++t) {
    const int gcol = nBase + 64 * wcol + 16 * t + lr;
#pragma unroll
    for (int r = 0; r < 8; ++r) {
      const int grow = mBase + 16 * wrow + 8 * hf + r;  // C/D: row = r + 8*half
      if (grow < p.M && gcol < p.N) {
        float v = acc[t][r] * p.alpha;
        if (biasp) v += biasp[gcol];
        if (p.act == 1) v = gelu_erf(v);
        v *= rs;
        if (p.addend) v += p.addend[(long)grow * p.N + gcol];
        if (p.store_img) p.C[cOff + (long)gcol * p.ldc + grow] = v;
        else             p.C[cOff + (long)grow * p.ldc + gcol] = v;
      }
    }
  }
}

// ---------------- split-K reduction + epilogue ----------------
__global__ void reduce_epi(const float* part, int nz, long chunkStride,
                           float* dst, const float* bias, int act,
                           int total, int N) {
  const int i = blockIdx.x * 256 + threadIdx.x;
  if (i >= total) return;
  float s = 0.f;
  for (int zc = 0; zc < nz; ++zc) s += part[(long)zc * chunkStride + i];
  if (bias) s += bias[i % N];
  if (act == 1) s = gelu_erf(s);
  dst[i] = s;
}

// ---------------- softmax over 196 (scaled) ----------------
__global__ void softmax196(float* att, float scale) {
  __shared__ float red[256];
  const int t = threadIdx.x;
  float* p = att + (long)blockIdx.x * 196;
  float v = (t < 196) ? p[t] * scale : -3.0e38f;
  red[t] = v; __syncthreads();
  for (int o = 128; o > 0; o >>= 1) { if (t < o) red[t] = fmaxf(red[t], red[t + o]); __syncthreads(); }
  const float m = red[0]; __syncthreads();
  float e = (t < 196) ? expf(v - m) : 0.f;
  red[t] = e; __syncthreads();
  for (int o = 128; o > 0; o >>= 1) { if (t < o) red[t] += red[t + o]; __syncthreads(); }
  if (t < 196) p[t] = e / red[0];
}

// z[b][d][s] = reimg[b][d][s] + y[b][s][d]
__global__ void add_transpose(const float* y, const float* reimg, float* z, int total) {
  const int i = blockIdx.x * 256 + threadIdx.x;
  if (i >= total) return;
  const int s = i % 196; const int bd = i / 196;
  const int d = bd % 256; const int b = bd / 256;
  z[i] = reimg[i] + y[((long)b * 196 + s) * 256 + d];
}

__global__ void add_inplace(float* a, const float* b, int total) {
  const int i = blockIdx.x * 256 + threadIdx.x;
  if (i < total) a[i] += b[i];
}

// batch-norm batch stats over (B=64, S=196) per channel, biased var
__global__ void bn_reduce(const float* x, int C, float* mean, float* rstd) {
  __shared__ float s1[256], s2[256];
  const int c = blockIdx.x, t = threadIdx.x;
  float sum = 0.f, ss = 0.f;
  for (int i = t; i < 64 * 196; i += 256) {
    const int b = i / 196, s = i - b * 196;
    const float v = x[((long)b * C + c) * 196 + s];
    sum += v; ss += v * v;
  }
  s1[t] = sum; s2[t] = ss; __syncthreads();
  for (int o = 128; o > 0; o >>= 1) {
    if (t < o) { s1[t] += s1[t + o]; s2[t] += s2[t + o]; }
    __syncthreads();
  }
  if (t == 0) {
    const float m = s1[0] * (1.f / (64.f * 196.f));
    const float var = s2[0] * (1.f / (64.f * 196.f)) - m * m;
    mean[c] = m; rstd[c] = rsqrtf(var + 1e-5f);
  }
}

__global__ void bn_apply(const float* x, const float* mean, const float* rstd,
                         const float* g, const float* bb, int C,
                         float* out_img, float* out_rows, int total) {
  const int i = blockIdx.x * 256 + threadIdx.x;
  if (i >= total) return;
  const int s = i % 196; const int bc = i / 196;
  const int c = bc % C; const int b = bc / C;
  const float v = (x[i] - mean[c]) * rstd[c] * g[c] + bb[c];
  out_img[i] = v;
  if (out_rows) out_rows[((long)b * 196 + s) * C + c] = v;
}

// rd2 (8 logits) + top-1 argmax + raw-logit gate, one block per sample
__global__ void router_head(const float* r4, const float* w, const float* bias,
                            int* pos, float* gate) {
  __shared__ float red[256 * 8];
  const int b = blockIdx.x, t = threadIdx.x;
  float acc[8];
#pragma unroll
  for (int e = 0; e < 8; ++e) acc[e] = 0.f;
  for (int k = t; k < 3136; k += 256) {
    const float xv = r4[(long)b * 3136 + k];
#pragma unroll
    for (int e = 0; e < 8; ++e) acc[e] += xv * w[e * 3136 + k];
  }
#pragma unroll
  for (int e = 0; e < 8; ++e) red[t * 8 + e] = acc[e];
  __syncthreads();
  for (int o = 128; o > 0; o >>= 1) {
    if (t < o) {
#pragma unroll
      for (int e = 0; e < 8; ++e) red[t * 8 + e] += red[(t + o) * 8 + e];
    }
    __syncthreads();
  }
  if (t == 0) {
    float best = -3.0e38f; int bi = 0;
    for (int e = 0; e < 8; ++e) {
      const float l = red[e] + bias[e];
      if (l > best) { best = l; bi = e; }
    }
    pos[b] = bi; gate[b] = best;
  }
}

// ---------------- host orchestration ----------------
static inline GemmParams GP() {
  GemmParams p{};
  p.alpha = 1.f;
  return p;
}

static void launch_gemm(const GemmParams& p, int Z, hipStream_t s) {
  dim3 g((unsigned)((p.N + 127) / 128), (unsigned)((p.M + 63) / 64), (unsigned)Z);
  const int key = p.amode * 2 + p.bmode;
  switch (key) {
    case 0: gemm_k<0, 0><<<g, 256, 0, s>>>(p); break;
    case 1: gemm_k<0, 1><<<g, 256, 0, s>>>(p); break;
    case 2: gemm_k<1, 0><<<g, 256, 0, s>>>(p); break;
    default: gemm_k<2, 0><<<g, 256, 0, s>>>(p); break;
  }
}

extern "C" void kernel_launch(void* const* d_in, const int* in_sizes, int n_in,
                              void* d_out, int out_size, void* d_ws, size_t ws_size,
                              hipStream_t stream) {
  (void)in_sizes; (void)n_in; (void)out_size; (void)ws_size;
  const float* x        = (const float*)d_in[0];
  const float* patch_w  = (const float*)d_in[1];
  const float* patch_b  = (const float*)d_in[2];
  const float* pos_emb  = (const float*)d_in[3];
  const float* wo_w     = (const float*)d_in[4];
  const float* wo_b     = (const float*)d_in[5];
  const float* bn1_g    = (const float*)d_in[6];
  const float* bn1_b    = (const float*)d_in[7];
  const float* rc1_w    = (const float*)d_in[8];
  const float* rc1_b    = (const float*)d_in[9];
  const float* rc2_w    = (const float*)d_in[10];
  const float* rc2_b    = (const float*)d_in[11];
  const float* rc3_w    = (const float*)d_in[12];
  const float* rc3_b    = (const float*)d_in[13];
  const float* rd1_w    = (const float*)d_in[14];
  const float* rd1_b    = (const float*)d_in[15];
  const float* rd2_w    = (const float*)d_in[16];
  const float* rd2_b    = (const float*)d_in[17];
  const float* exp_w1   = (const float*)d_in[18];
  const float* exp_b1   = (const float*)d_in[19];
  const float* exp_w2   = (const float*)d_in[20];
  const float* exp_b2   = (const float*)d_in[21];
  const float* shard_w  = (const float*)d_in[22];
  const float* shard_b  = (const float*)d_in[23];
  const float* bn2_g    = (const float*)d_in[24];
  const float* bn2_b    = (const float*)d_in[25];
  const float* od1_w    = (const float*)d_in[26];
  const float* od1_b    = (const float*)d_in[27];
  const float* od2_w    = (const float*)d_in[28];
  const float* od2_b    = (const float*)d_in[29];
  float* out = (float*)d_out;

  const int Bn = 64, S = 196, D = 256, HE = 256;
  const long yN = (long)Bn * S * D;  // 3,211,264 floats

  float* ws = (float*)d_ws;
  long o = 0;
  const long o_y      = o; o += yN;                  // reused as bn2 output o2
  const long o_att    = o; o += (long)Bn * S * S;
  const long o_re     = o; o += yN;                  // reused as shard out1
  const long o_reimg  = o; o += yN;
  const long o_z1     = o; o += yN;                  // reused as expert h1
  const long o_outimg = o; o += yN;
  const long o_outrow = o; o += yN;
  const long o_r1     = o; o += (long)Bn * 128 * S;
  const long o_r2     = o; o += (long)Bn * 64 * S;
  const long o_r3     = o; o += (long)Bn * 32 * S;
  const long o_r4     = o; o += (long)Bn * 3136;
  const long o_h2     = o; o += yN;
  const long o_o3     = o; o += (long)Bn * 1024;
  const long o_part   = o; o += (long)14 * Bn * 3136;  // split-K partials (max user: rd1)
  const long o_mean   = o; o += 256;
  const long o_rstd   = o; o += 256;
  const long o_gate   = o; o += 64;
  const long o_posi   = o; o += 64;
  int* d_pos = (int*)(ws + o_posi);
  const long o_h1 = o_z1, o_out1 = o_re, o_o2 = o_y;  // buffer reuse (sequential stream)

  // 1) patch embed + pos_emb -> y[B,S,D]
  { GemmParams p = GP();
    p.A = x; p.strideA = (long)3 * 224 * 224; p.amode = 2;
    p.B = patch_w; p.ldb = 768; p.bmode = 0;
    p.bias = patch_b; p.addend = pos_emb;
    p.C = ws + o_y; p.strideC = (long)S * D; p.ldc = D;
    p.M = S; p.N = D; p.K = 768;
    launch_gemm(p, Bn, stream); }

  // 2) scores = y @ y^T -> att[B,S,S]
  { GemmParams p = GP();
    p.A = ws + o_y; p.lda = D; p.strideA = (long)S * D; p.amode = 0;
    p.B = ws + o_y; p.ldb = D; p.strideB = (long)S * D; p.bmode = 0;  // B as [N=S,K=D]
    p.C = ws + o_att; p.ldc = S; p.strideC = (long)S * S;
    p.M = S; p.N = S; p.K = D;
    launch_gemm(p, Bn, stream); }

  // 3) softmax(att / sqrt(14))
  softmax196<<<Bn * S, 256, 0, stream>>>(ws + o_att, 0.26726124191242440f);

  // 4) re = att @ y -> [B,S,D]
  { GemmParams p = GP();
    p.A = ws + o_att; p.lda = S; p.strideA = (long)S * S; p.amode = 0;
    p.B = ws + o_y; p.ldb = D; p.strideB = (long)S * D; p.bmode = 1;  // B as [K=S,N=D]
    p.C = ws + o_re; p.ldc = D; p.strideC = (long)S * D;
    p.M = S; p.N = D; p.K = S;
    launch_gemm(p, Bn, stream); }

  // 5) re @ wo^T + wo_b -> re_img [B,D,S]
  { GemmParams p = GP();
    p.A = ws + o_re; p.lda = D; p.strideA = (long)S * D; p.amode = 0;
    p.B = wo_w; p.ldb = D; p.bmode = 0; p.bias = wo_b;
    p.C = ws + o_reimg; p.store_img = 1; p.ldc = S; p.strideC = (long)D * S;
    p.M = S; p.N = D; p.K = D;
    launch_gemm(p, Bn, stream); }

  // 6) z1 = y_img + re_img
  add_transpose<<<(unsigned)((yN + 255) / 256), 256, 0, stream>>>(
      ws + o_y, ws + o_reimg, ws + o_z1, (int)yN);

  // 7-8) bn1 -> out_img [B,D,S] + out_rows [B,S,D]
  bn_reduce<<<D, 256, 0, stream>>>(ws + o_z1, D, ws + o_mean, ws + o_rstd);
  bn_apply<<<(unsigned)((yN + 255) / 256), 256, 0, stream>>>(
      ws + o_z1, ws + o_mean, ws + o_rstd, bn1_g, bn1_b, D,
      ws + o_outimg, ws + o_outrow, (int)yN);

  // 9-11) router convs (implicit GEMM, GELU)
  { GemmParams p = GP();
    p.amode = 1; p.bmode = 0; p.act = 1; p.store_img = 1; p.ldc = S; p.M = S;
    p.A = ws + o_outimg; p.strideA = (long)D * S;
    p.B = rc1_w; p.ldb = 2304; p.bias = rc1_b;
    p.C = ws + o_r1; p.strideC = (long)128 * S; p.N = 128; p.K = 2304;
    launch_gemm(p, Bn, stream);
    p.A = ws + o_r1; p.strideA = (long)128 * S;
    p.B = rc2_w; p.ldb = 1152; p.bias = rc2_b;
    p.C = ws + o_r2; p.strideC = (long)64 * S; p.N = 64; p.K = 1152;
    launch_gemm(p, Bn, stream);
    p.A = ws + o_r2; p.strideA = (long)64 * S;
    p.B = rc3_w; p.ldb = 576; p.bias = rc3_b;
    p.C = ws + o_r3; p.strideC = (long)32 * S; p.N = 32; p.K = 576;
    launch_gemm(p, Bn, stream); }

  // 12) rd1 [64,6272]x[6272,3136], split-K (14 chunks of 448) + GELU reduce
  { GemmParams p = GP();
    p.A = ws + o_r3; p.lda = 6272; p.amode = 0;
    p.B = rd1_w; p.ldb = 6272; p.bmode = 0;
    p.C = ws + o_part; p.ldc = 3136; p.strideC = (long)Bn * 3136;
    p.M = Bn; p.N = 3136; p.K = 6272; p.splitk = 448;
    launch_gemm(p, 14, stream);
    reduce_epi<<<(unsigned)((Bn * 3136 + 255) / 256), 256, 0, stream>>>(
        ws + o_part, 14, (long)Bn * 3136, ws + o_r4, rd1_b, 1, Bn * 3136, 3136); }

  // 13) rd2 + argmax -> pos, gate
  router_head<<<Bn, 256, 0, stream>>>(ws + o_r4, rd2_w, rd2_b, d_pos, ws + o_gate);

  // 14-15) expert convs (per-sample weights); gate applied on conv2
  { GemmParams p = GP();
    p.amode = 1; p.bmode = 0; p.act = 1; p.store_img = 1; p.ldc = S; p.M = S;
    p.sel = d_pos; p.biasSel = 1;
    p.A = ws + o_reimg; p.strideA = (long)D * S;
    p.B = exp_w1; p.selStrideB = (long)HE * 2304; p.ldb = 2304; p.bias = exp_b1;
    p.C = ws + o_h1; p.strideC = (long)HE * S; p.N = HE; p.K = 2304;
    launch_gemm(p, Bn, stream);
    p.A = ws + o_h1; p.strideA = (long)HE * S;
    p.B = exp_w2; p.selStrideB = (long)HE * HE * 9; p.ldb = 2304; p.bias = exp_b2;
    p.rowscale = ws + o_gate;
    p.C = ws + o_h2; p.strideC = (long)HE * S; p.N = HE; p.K = 2304;
    launch_gemm(p, Bn, stream); }

  // 16) shard 1x1 conv on bn1-out -> out1 img
  { GemmParams p = GP();
    p.A = ws + o_outrow; p.lda = D; p.strideA = (long)S * D; p.amode = 0;
    p.B = shard_w; p.ldb = D; p.bmode = 0; p.bias = shard_b;
    p.C = ws + o_out1; p.store_img = 1; p.ldc = S; p.strideC = (long)HE * S;
    p.M = S; p.N = HE; p.K = D;
    launch_gemm(p, Bn, stream); }

  // 17) h2 += out1
  add_inplace<<<(unsigned)((yN + 255) / 256), 256, 0, stream>>>(
      ws + o_h2, ws + o_out1, (int)yN);

  // 18) bn2 -> o2 [64, 50176]
  bn_reduce<<<HE, 256, 0, stream>>>(ws + o_h2, HE, ws + o_mean, ws + o_rstd);
  bn_apply<<<(unsigned)((yN + 255) / 256), 256, 0, stream>>>(
      ws + o_h2, ws + o_mean, ws + o_rstd, bn2_g, bn2_b, HE,
      ws + o_o2, nullptr, (int)yN);

  // 19) od1 [64,50176]x[50176,1024], split-K (25 chunks of 2048) + GELU reduce
  { GemmParams p = GP();
    p.A = ws + o_o2; p.lda = 50176; p.amode = 0;
    p.B = od1_w; p.ldb = 50176; p.bmode = 0;
    p.C = ws + o_part; p.ldc = 1024; p.strideC = (long)Bn * 1024;
    p.M = Bn; p.N = 1024; p.K = 50176; p.splitk = 2048;
    launch_gemm(p, 25, stream);
    reduce_epi<<<(unsigned)((Bn * 1024 + 255) / 256), 256, 0, stream>>>(
        ws + o_part, 25, (long)Bn * 1024, ws + o_o3, od1_b, 1, Bn * 1024, 1024); }

  // 20) od2 [64,1024]x[1024,1000], split-K (4 chunks of 256) -> d_out
  { GemmParams p = GP();
    p.A = ws + o_o3; p.lda = 1024; p.amode = 0;
    p.B = od2_w; p.ldb = 1024; p.bmode = 0;
    p.C = ws + o_part; p.ldc = 1000; p.strideC = (long)Bn * 1000;
    p.M = Bn; p.N = 1000; p.K = 1024; p.splitk = 256;
    launch_gemm(p, 4, stream);
    reduce_epi<<<(unsigned)((Bn * 1000 + 255) / 256), 256, 0, stream>>>(
        ws + o_part, 4, (long)Bn * 1000, out, od2_b, 0, Bn * 1000, 1000); }
}
